// CriticScorer_51445118272147
// MI455X (gfx1250) — matrix operand order repaired and checked by reading, not compiled, not run on hardware
//
#include <hip/hip_runtime.h>

#define NS 16384   // num sentences (rows of attn_mem)
#define DD 1024    // input dim
#define HH 1024    // hidden dim

typedef __attribute__((ext_vector_type(16))) __bf16 v16bf;
typedef __attribute__((ext_vector_type(8)))  float  v8f;
typedef __attribute__((ext_vector_type(4)))  float  f4;

// ---------- helpers ----------
static __device__ __forceinline__ unsigned short f2bf_bits(float f) {
    unsigned u = __builtin_bit_cast(unsigned, f);
    return (unsigned short)((u + 0x7FFFu + ((u >> 16) & 1u)) >> 16);  // RNE
}
static __device__ __forceinline__ float bf_bits2f(unsigned short s) {
    unsigned u = ((unsigned)s) << 16;
    return __builtin_bit_cast(float, u);
}
static __device__ __forceinline__ __bf16 bits2bf(unsigned short s) {
    return __builtin_bit_cast(__bf16, s);
}
static __device__ __forceinline__ float wave_sum(float v) {
#pragma unroll
    for (int o = 16; o > 0; o >>= 1) v += __shfl_xor(v, o, 32);
    return v;
}
static __device__ __forceinline__ float sigmoidf_fast(float x) {
    return 1.0f / (1.0f + __expf(-x));
}

// ---------- weight transpose: Wt[n*K + k] = W[k*H + n]  (1024x1024) ----------
__global__ void transpose_kernel(const float* __restrict__ W, float* __restrict__ Wt) {
    int tid = blockIdx.x * blockDim.x + threadIdx.x;
    int k = tid >> 10, n = tid & 1023;
    Wt[(size_t)n * HH + k] = W[(size_t)k * HH + n];
}

// ---------- dual feature GEMM: featA = A@Wa, featH = A@Wh via bf16x3 WMMA ----------
// One wave -> one 16x16 tile of BOTH outputs (A fragment reused).
// Block: 256 thr = 8 waves covering 128 consecutive columns of one 16-row M tile.
__global__ void feat_gemm_kernel(const float* __restrict__ A,
                                 const float* __restrict__ WtA,   // [HH x DD] transposed
                                 const float* __restrict__ WtH,   // [HH x DD] transposed
                                 float* __restrict__ featA,
                                 float* __restrict__ featH) {
    const int lane = threadIdx.x & 31;
    const int wave = threadIdx.x >> 5;
    const int mt = blockIdx.x >> 3;          // 1024 M tiles
    const int ng = blockIdx.x & 7;           // 8 column groups of 128
    const int nt = ng * 8 + wave;            // 64 N tiles total
    const int m0 = mt * 16, n0 = nt * 16;
    const int half = lane >> 4;
    const int mr = lane & 15;

    // A 16x32 layout: element i of v16bf <-> K = i + half*16, M = lane&15.
    const float* arow  = A   + (size_t)(m0 + mr) * DD + half * 16;
    // B (KxN) layout: element i of v16bf <-> K = i + half*16, N = lane&15.
    const float* browA = WtA + (size_t)(n0 + mr) * DD + half * 16;
    const float* browH = WtH + (size_t)(n0 + mr) * DD + half * 16;

    v8f acc0 = {}; v8f acc1 = {};
    for (int kt = 0; kt < DD / 32; ++kt) {
        const int kb = kt * 32;
        // A fragment: 16 contiguous floats
        const f4* ap = (const f4*)(arow + kb);
        f4 a0 = ap[0], a1 = ap[1], a2 = ap[2], a3 = ap[3];
        float av[16];
#pragma unroll
        for (int i = 0; i < 4; ++i) {
            av[i] = a0[i]; av[4 + i] = a1[i]; av[8 + i] = a2[i]; av[12 + i] = a3[i];
        }
        // B fragments: 16 contiguous floats each
        const f4* bpA = (const f4*)(browA + kb);
        const f4* bpH = (const f4*)(browH + kb);
        float bvA[16], bvH[16];
#pragma unroll
        for (int i = 0; i < 4; ++i) {
            f4 t0 = bpA[i], t1 = bpH[i];
#pragma unroll
            for (int j = 0; j < 4; ++j) { bvA[i * 4 + j] = t0[j]; bvH[i * 4 + j] = t1[j]; }
        }
        // hi/lo bf16 split (error compensation)
        v16bf ahi, alo, bhiA, bloA, bhiH, bloH;
#pragma unroll
        for (int i = 0; i < 16; ++i) {
            unsigned short h;
            h = f2bf_bits(av[i]);  ahi[i]  = bits2bf(h); alo[i]  = bits2bf(f2bf_bits(av[i]  - bf_bits2f(h)));
            h = f2bf_bits(bvA[i]); bhiA[i] = bits2bf(h); bloA[i] = bits2bf(f2bf_bits(bvA[i] - bf_bits2f(h)));
            h = f2bf_bits(bvH[i]); bhiH[i] = bits2bf(h); bloH[i] = bits2bf(f2bf_bits(bvH[i] - bf_bits2f(h)));
        }
        acc0 = __builtin_amdgcn_wmma_f32_16x16x32_bf16(false, ahi, false, bhiA, (short)0, acc0, false, false);
        acc0 = __builtin_amdgcn_wmma_f32_16x16x32_bf16(false, alo, false, bhiA, (short)0, acc0, false, false);
        acc0 = __builtin_amdgcn_wmma_f32_16x16x32_bf16(false, ahi, false, bloA, (short)0, acc0, false, false);
        acc1 = __builtin_amdgcn_wmma_f32_16x16x32_bf16(false, ahi, false, bhiH, (short)0, acc1, false, false);
        acc1 = __builtin_amdgcn_wmma_f32_16x16x32_bf16(false, alo, false, bhiH, (short)0, acc1, false, false);
        acc1 = __builtin_amdgcn_wmma_f32_16x16x32_bf16(false, ahi, false, bloH, (short)0, acc1, false, false);
    }
    // C layout: VGPR r -> row m0 + r + half*8, col n0 + (lane&15)
#pragma unroll
    for (int r = 0; r < 8; ++r) {
        size_t idx = (size_t)(m0 + half * 8 + r) * HH + (n0 + mr);
        featA[idx] = acc0[r];
        featH[idx] = acc1[r];
    }
}

// ---------- init: load recurrent state ----------
__global__ void init_kernel(const float* __restrict__ ih, const float* __restrict__ ic,
                            const float* __restrict__ ii,
                            float* __restrict__ h0, float* __restrict__ c, float* __restrict__ x) {
    int t = blockIdx.x * blockDim.x + threadIdx.x;
    if (t < HH) { h0[t] = ih[t]; c[t] = ic[t]; x[t] = ii[t]; }
}

// ---------- LSTM cell: one wave per hidden row, all 4 gates ----------
__global__ void lstm_kernel(const float* __restrict__ x, const float* __restrict__ hin,
                            float* __restrict__ c,
                            const float* __restrict__ Wih, const float* __restrict__ Whh,
                            const float* __restrict__ bih, const float* __restrict__ bhh,
                            float* __restrict__ hout) {
    const int lane = threadIdx.x & 31;
    const int j = blockIdx.x * (blockDim.x >> 5) + (threadIdx.x >> 5);
    const float* wi0 = Wih + (size_t)j * DD;
    const float* wi1 = Wih + (size_t)(j + HH) * DD;
    const float* wi2 = Wih + (size_t)(j + 2 * HH) * DD;
    const float* wi3 = Wih + (size_t)(j + 3 * HH) * DD;
    const float* wh0 = Whh + (size_t)j * HH;
    const float* wh1 = Whh + (size_t)(j + HH) * HH;
    const float* wh2 = Whh + (size_t)(j + 2 * HH) * HH;
    const float* wh3 = Whh + (size_t)(j + 3 * HH) * HH;
    float ai = 0.f, af = 0.f, ag = 0.f, ao = 0.f;
    for (int i = 0; i < DD / 32; ++i) {
        int k = i * 32 + lane;
        float xv = x[k], hv = hin[k];
        ai += wi0[k] * xv + wh0[k] * hv;
        af += wi1[k] * xv + wh1[k] * hv;
        ag += wi2[k] * xv + wh2[k] * hv;
        ao += wi3[k] * xv + wh3[k] * hv;
    }
    ai = wave_sum(ai); af = wave_sum(af); ag = wave_sum(ag); ao = wave_sum(ao);
    if (lane == 0) {
        float gi = sigmoidf_fast(ai + bih[j] + bhh[j]);
        float gf = sigmoidf_fast(af + bih[j + HH] + bhh[j + HH]);
        float gg = tanhf(ag + bih[j + 2 * HH] + bhh[j + 2 * HH]);
        float go = sigmoidf_fast(ao + bih[j + 3 * HH] + bhh[j + 3 * HH]);
        float cn = gf * c[j] + gi * gg;
        c[j] = cn;
        hout[j] = go * tanhf(cn);
    }
}

// ---------- small matvec: out[j] = sum_k q[k] * W[k*HH + j] ----------
__global__ void matvec_kernel(const float* __restrict__ q, const float* __restrict__ W,
                              float* __restrict__ out) {
    int j = blockIdx.x * blockDim.x + threadIdx.x;
    float s = 0.f;
    for (int k = 0; k < HH; ++k) s += q[k] * W[(size_t)k * HH + j];
    out[j] = s;
}

// ---------- attention logits: e[n] = sum_k tanh(feat[n,k]+qw[k]) * v[k] ----------
__global__ void attn_e_kernel(const float* __restrict__ feat, const float* __restrict__ qw,
                              const float* __restrict__ v, float* __restrict__ e) {
    const int lane = threadIdx.x & 31;
    const int n = blockIdx.x * (blockDim.x >> 5) + (threadIdx.x >> 5);
    const float* row = feat + (size_t)n * HH;
    float acc = 0.f;
    for (int i = 0; i < HH / 32; ++i) {
        int k = i * 32 + lane;
        acc += tanhf(row[k] + qw[k]) * v[k];
    }
    acc = wave_sum(acc);
    if (lane == 0) e[n] = acc;
}

// ---------- softmax over NS (single block) ----------
__global__ void softmax_kernel(const float* __restrict__ e, float* __restrict__ a) {
    __shared__ float red[256];
    const int t = threadIdx.x;
    float mx = -3.4e38f;
    for (int i = t; i < NS; i += 256) mx = fmaxf(mx, e[i]);
    red[t] = mx; __syncthreads();
    for (int s = 128; s > 0; s >>= 1) { if (t < s) red[t] = fmaxf(red[t], red[t + s]); __syncthreads(); }
    mx = red[0]; __syncthreads();
    float sm = 0.f;
    for (int i = t; i < NS; i += 256) sm += __expf(e[i] - mx);
    red[t] = sm; __syncthreads();
    for (int s = 128; s > 0; s >>= 1) { if (t < s) red[t] += red[t + s]; __syncthreads(); }
    float inv = 1.f / red[0];
    for (int i = t; i < NS; i += 256) a[i] = __expf(e[i] - mx) * inv;
}

// ---------- weighted sum stage 1: per-256-row-block partial ctx ----------
__global__ void wsum_partial_kernel(const float* __restrict__ a, const float* __restrict__ mem,
                                    float* __restrict__ partial) {
    const int t = threadIdx.x;
    const int b = blockIdx.x;
    const int n0 = b * (NS / 64);
    float a0 = 0.f, a1 = 0.f, a2 = 0.f, a3 = 0.f;
    for (int n = 0; n < NS / 64; ++n) {
        float w = a[n0 + n];
        const float* row = mem + (size_t)(n0 + n) * HH;
        a0 += w * row[t];
        a1 += w * row[t + 256];
        a2 += w * row[t + 512];
        a3 += w * row[t + 768];
    }
    partial[b * HH + t]       = a0;
    partial[b * HH + t + 256] = a1;
    partial[b * HH + t + 512] = a2;
    partial[b * HH + t + 768] = a3;
}

// ---------- weighted sum stage 2 ----------
__global__ void wsum_final_kernel(const float* __restrict__ partial, float* __restrict__ out) {
    int col = blockIdx.x * blockDim.x + threadIdx.x;
    float s = 0.f;
    for (int b = 0; b < 64; ++b) s += partial[b * HH + col];
    out[col] = s;
}

// ---------- score: scores[t] = dot(out, score_w) + score_b ----------
__global__ void score_kernel(const float* __restrict__ outv, const float* __restrict__ sw,
                             const float* __restrict__ sb, float* __restrict__ dst) {
    __shared__ float red[256];
    const int t = threadIdx.x;
    float s = 0.f;
    for (int i = t; i < DD; i += 256) s += outv[i] * sw[i];
    red[t] = s; __syncthreads();
    for (int k = 128; k > 0; k >>= 1) { if (t < k) red[t] += red[t + k]; __syncthreads(); }
    if (t == 0) dst[0] = red[0] + sb[0];
}

extern "C" void kernel_launch(void* const* d_in, const int* in_sizes, int n_in,
                              void* d_out, int out_size, void* d_ws, size_t ws_size,
                              hipStream_t stream) {
    (void)in_sizes; (void)n_in; (void)ws_size;
    const float* attn_mem = (const float*)d_in[0];
    // d_in[1] = num_outputs (device int) — trip count equals out_size, use that.
    const float* init_h  = (const float*)d_in[2];
    const float* init_c  = (const float*)d_in[3];
    const float* init_i  = (const float*)d_in[4];
    const float* W_ih    = (const float*)d_in[5];
    const float* W_hh    = (const float*)d_in[6];
    const float* b_ih    = (const float*)d_in[7];
    const float* b_hh    = (const float*)d_in[8];
    const float* attn_wm = (const float*)d_in[9];
    const float* attn_wq = (const float*)d_in[10];
    const float* attn_v  = (const float*)d_in[11];
    const float* hop_wm  = (const float*)d_in[12];
    const float* hop_wq  = (const float*)d_in[13];
    const float* hop_v   = (const float*)d_in[14];
    const float* score_w = (const float*)d_in[15];
    const float* score_b = (const float*)d_in[16];
    float* out = (float*)d_out;

    // workspace layout (floats)
    float* ws = (float*)d_ws;
    size_t off = 0;
    float* WtA   = ws + off; off += (size_t)DD * HH;
    float* WtH   = ws + off; off += (size_t)DD * HH;
    float* featA = ws + off; off += (size_t)NS * HH;
    float* featH = ws + off; off += (size_t)NS * HH;
    float* e     = ws + off; off += NS;
    float* aw    = ws + off; off += NS;
    float* qw    = ws + off; off += HH;
    float* ctx   = ws + off; off += HH;
    float* xb    = ws + off; off += DD;
    float* hA    = ws + off; off += HH;
    float* hB    = ws + off; off += HH;
    float* cb    = ws + off; off += HH;
    float* part  = ws + off; off += (size_t)64 * HH;

    transpose_kernel<<<4096, 256, 0, stream>>>(attn_wm, WtA);
    transpose_kernel<<<4096, 256, 0, stream>>>(hop_wm, WtH);
    init_kernel<<<4, 256, 0, stream>>>(init_h, init_c, init_i, hA, cb, xb);
    feat_gemm_kernel<<<8192, 256, 0, stream>>>(attn_mem, WtA, WtH, featA, featH);

    float* hin = hA;
    float* hout = hB;
    const int T = out_size;
    for (int t = 0; t < T; ++t) {
        lstm_kernel<<<128, 256, 0, stream>>>(xb, hin, cb, W_ih, W_hh, b_ih, b_hh, hout);
        // hop attention (q = h_new) over hop_feat
        matvec_kernel<<<4, 256, 0, stream>>>(hout, hop_wq, qw);
        attn_e_kernel<<<2048, 256, 0, stream>>>(featH, qw, hop_v, e);
        softmax_kernel<<<1, 256, 0, stream>>>(e, aw);
        wsum_partial_kernel<<<64, 256, 0, stream>>>(aw, featH, part);
        wsum_final_kernel<<<4, 256, 0, stream>>>(part, ctx);
        // output attention (q = ctx) over attn_mem / attn_feat
        matvec_kernel<<<4, 256, 0, stream>>>(ctx, attn_wq, qw);
        attn_e_kernel<<<2048, 256, 0, stream>>>(featA, qw, attn_v, e);
        softmax_kernel<<<1, 256, 0, stream>>>(e, aw);
        wsum_partial_kernel<<<64, 256, 0, stream>>>(aw, attn_mem, part);
        wsum_final_kernel<<<4, 256, 0, stream>>>(part, xb);   // out -> next x
        score_kernel<<<1, 256, 0, stream>>>(xb, score_w, score_b, out + t);
        float* tmp = hin; hin = hout; hout = tmp;
    }
}
